// GATDQNBase_12266426597750
// MI455X (gfx1250) — matrix operand order repaired and checked by reading, not compiled, not run on hardware
//
#include <hip/hip_runtime.h>
#include <hip/hip_bf16.h>

// ---------------------------------------------------------------------------
// GAT-DQN on MI455X (gfx1250, wave32, WMMA).
// B=8 N=2048 Fin=16 E=64 H=4 D=16 HID=128 A=4.
//
// Roofline: adjacency (134MB) is the only big input -> bit-pack once to 4.2MB
// (L2-resident for both layers). Attention is fused flash-style (scores never
// hit HBM), 32 keys per inner iteration: 2x QK^T wmma + 1x full-K=32 PV wmma.
// All activations f16, accumulate f32 via v_wmma_f32_16x16x32_f16.
// Q/out projections are algebraically folded (affine compositions).
// ---------------------------------------------------------------------------

typedef __attribute__((ext_vector_type(16))) _Float16 v16h;
typedef __attribute__((ext_vector_type(8)))  float    v8f;
typedef unsigned int u32;

#define BB 8
#define NN 2048
#define EE 64
#define HH 4
#define MROWS (BB*NN)        // 16384 token rows
#define WPR  (NN/32)         // mask words per row = 64

__device__ __forceinline__ v8f wmma_f16(v16h a, v16h b, v8f c) {
  // D = A(16x32,f16) * B(32x16,f16) + C(16x16,f32)
  return __builtin_amdgcn_wmma_f32_16x16x32_f16(
      /*neg_a=*/false, a, /*neg_b=*/false, b,
      /*c_mod=*/(short)0, c, /*reuse_a=*/false, /*reuse_b=*/false);
}

// A-layout K index for element e (16-bit A matrix 16x32, wave32):
//  lanes 0-15: e0..7 -> K=e, e8..15 -> K=16+(e-8)
//  lanes16-31: e0..7 -> K=8+e, e8..15 -> K=24+(e-8)
__device__ __forceinline__ int a_kidx(int e, int half) {
  return (e < 8) ? (e + 8*half) : (16 + (e-8) + 8*half);
}

// ---------------- input scaling: x * scale(col) -> f16 ---------------------
__global__ __launch_bounds__(256) void scale_x(const float* __restrict__ x,
                                               _Float16* __restrict__ x16) {
  int idx = blockIdx.x*256 + threadIdx.x;            // over MROWS*16
  int col = idx & 15;
  // QUEUE_IDX = {0..3,6..13} -> 1/50 ; col 5 -> 1/300 ; cols 4,14,15 -> 1
  float s = (col == 5) ? (1.f/300.f) : ((col == 4 || col >= 14) ? 1.f : 0.02f);
  x16[idx] = (_Float16)(x[idx] * s);
}

// ------------- pack adjacency: bit=1 means BLOCKED (adj==0 && k!=q) --------
__global__ __launch_bounds__(256) void pack_mask(const float* __restrict__ adj,
                                                 u32* __restrict__ mp) {
  size_t idx = (size_t)blockIdx.x*256 + threadIdx.x; // word index, B*N*WPR
  size_t bq = idx / WPR;
  int w = (int)(idx % WPR);
  int q = (int)(bq % NN);
  const float* arow = adj + bq*NN + (size_t)w*32;
  u32 bits = 0;
  #pragma unroll 4
  for (int i = 0; i < 32; ++i) {
    int key = w*32 + i;
    bool blocked = (arow[i] == 0.f) && (key != q);
    bits |= (u32)blocked << i;
  }
  mp[idx] = bits;
}

// ---------------- fold all weights, transpose for B-operand ----------------
// f16 offsets (elements):
//  WqT0=0 WkT0=1024 WvT0=2048 (64 x 16 each)
//  WqT1=3072 WkT1=7168 WvT1=11264 (64 x 64 each)
//  WoT0=15360 WoT1=19456 (64 x 64)
//  W1T=23552 (128x64)  W2T=31744 (128x128)  W3T=48128 (16x128, rows>=4 zero)
// f32 bias offsets: bq0=0 bk0=64 bv0=128 bo0=192 | +256 for layer1
//  b1=512(128) b2=640(128) b3=768(16)
__global__ __launch_bounds__(256) void fold_weights(
    const float* wq0, const float* wk0, const float* wv0,
    const float* inw0, const float* inb0,
    const float* mow0, const float* mob0, const float* opw0, const float* opb0,
    const float* wq1, const float* wk1, const float* wv1,
    const float* inw1, const float* inb1,
    const float* mow1, const float* mob1, const float* opw1, const float* opb1,
    const float* w1, const float* b1, const float* w2, const float* b2,
    const float* w3, const float* b3,
    _Float16* wf, float* bf)
{
  const int t = threadIdx.x;
  for (int lyr = 0; lyr < 2; ++lyr) {
    const int fin = lyr ? 64 : 16;
    const float* proj[3] = { lyr ? wq1 : wq0, lyr ? wk1 : wk0, lyr ? wv1 : wv0 };
    const float* inw = lyr ? inw1 : inw0;
    const float* inb = lyr ? inb1 : inb0;
    for (int p = 0; p < 3; ++p) {
      _Float16* dst = wf + (lyr ? (3072 + p*4096) : (p*1024));
      const float* W = proj[p];
      for (int idx = t; idx < 64*fin; idx += 256) {
        int n = idx / fin, f = idx % fin;
        float s = 0.f;
        for (int j = 0; j < 64; ++j) s += W[f*64 + j] * inw[j*192 + p*64 + n];
        dst[idx] = (_Float16)s;                      // WpT[n][f]
      }
      for (int n = t; n < 64; n += 256) bf[lyr*256 + p*64 + n] = inb[p*64 + n];
    }
    const float* mow = lyr ? mow1 : mow0;
    const float* mob = lyr ? mob1 : mob0;
    const float* opw = lyr ? opw1 : opw0;
    const float* opb = lyr ? opb1 : opb0;
    _Float16* dsto = wf + (lyr ? 19456 : 15360);
    for (int idx = t; idx < 4096; idx += 256) {
      int n = idx >> 6, k = idx & 63;
      float s = 0.f;
      for (int j = 0; j < 64; ++j) s += mow[k*64 + j] * opw[j*64 + n];
      dsto[idx] = (_Float16)s;                       // WoT[n][k]
    }
    for (int n = t; n < 64; n += 256) {
      float s = opb[n];
      for (int j = 0; j < 64; ++j) s += mob[j] * opw[j*64 + n];
      bf[lyr*256 + 192 + n] = s;
    }
  }
  for (int idx = t; idx < 128*64; idx += 256) {
    int n = idx / 64, k = idx % 64;
    wf[23552 + idx] = (_Float16)w1[k*128 + n];
  }
  for (int idx = t; idx < 128*128; idx += 256) {
    int n = idx >> 7, k = idx & 127;
    wf[31744 + idx] = (_Float16)w2[k*128 + n];
  }
  for (int idx = t; idx < 16*128; idx += 256) {
    int n = idx >> 7, k = idx & 127;
    wf[48128 + idx] = (n < 4) ? (_Float16)w3[k*4 + n] : (_Float16)0.f;
  }
  for (int n = t; n < 128; n += 256) { bf[512 + n] = b1[n]; bf[640 + n] = b2[n]; }
  for (int n = t; n < 16; n += 256)  bf[768 + n] = (n < 4) ? b3[n] : 0.f;
}

// ---------------- generic 16x16-tile WMMA GEMM -----------------------------
// C[M,N] = act(A[M,K](f16,row-major) * WT[N,K]^T (f16, pre-transposed) + bias)
// OMODE 0: f16 row-major (ldo)   1: f32 row-major, cols < ncmax only
//       2: scatter into Vt[b][h][d][NN]
template<int K, int OMODE, bool RELU>
__global__ __launch_bounds__(256) void gemm16(
    const _Float16* __restrict__ A, const _Float16* __restrict__ WT,
    const float* __restrict__ bias, void* __restrict__ Out,
    int M, int NT, int ldo, int ncmax)
{
  const int lane = threadIdx.x & 31;
  const int half = lane >> 4, r = lane & 15;
  int wid = blockIdx.x * (blockDim.x >> 5) + (threadIdx.x >> 5);
  int tiles = (M >> 4) * NT;
  if (wid >= tiles) return;                 // wave-uniform: EXEC stays full
  int mt = wid / NT, nt = wid % NT;
  const int col = nt*16 + r;

  v8f acc = {};
  for (int k0 = 0; k0 < K; k0 += 32) {
    v16h a, b;
    const _Float16* arow = A + (size_t)(mt*16 + r) * K;
    #pragma unroll
    for (int e = 0; e < 16; ++e) {
      int ka = k0 + e + 16*half;
      a[e] = (ka < K) ? arow[ka] : (_Float16)0.f;
    }
    const _Float16* wrow = WT + (size_t)col * K;
    #pragma unroll
    for (int e = 0; e < 16; ++e) {          // B layout: K = e + 16*half
      int kb = k0 + e + 16*half;
      b[e] = (kb < K) ? wrow[kb] : (_Float16)0.f;
    }
    acc = wmma_f16(a, b, acc);
  }
  float bvv = bias[col];
  #pragma unroll
  for (int e = 0; e < 8; ++e) {             // C layout: row = e+8*half, col = r
    int row = mt*16 + e + 8*half;
    float v = acc[e] + bvv;
    if (RELU) v = fmaxf(v, 0.f);
    if (OMODE == 0) {
      ((_Float16*)Out)[(size_t)row*ldo + col] = (_Float16)v;
    } else if (OMODE == 1) {
      if (col < ncmax) ((float*)Out)[(size_t)row*ldo + col] = v;
    } else {
      int bb = row >> 11, nq = row & (NN-1);
      int hh = col >> 4, dd = col & 15;
      ((_Float16*)Out)[(((size_t)(bb*HH + hh))*16 + dd)*NN + nq] = (_Float16)v;
    }
  }
}

// ---------------- fused flash attention ------------------------------------
// One wave per (b, h, 16-row q-tile); 32 keys per iteration (one mask word).
// Online softmax; -1e30 mask surrogate (garbage from fully-masked early tiles
// is flushed by corr=exp(old-new)=0; diagonal is always unmasked so every row
// gets a real max eventually).
__global__ __launch_bounds__(128) void attn_kernel(
    const _Float16* __restrict__ Q, const _Float16* __restrict__ Km,
    const _Float16* __restrict__ Vt, const u32* __restrict__ maskp,
    _Float16* __restrict__ O)
{
  __shared__ _Float16 psh[4][16][32];       // P staged as 16x32 A-tile
  const int lane = threadIdx.x & 31;
  const int half = lane >> 4, r = lane & 15;
  const int w = threadIdx.x >> 5;
  int gid = blockIdx.x*4 + w;               // B*H*(N/16) = 4096 total
  int b  = gid >> 9;                        // H*(N/16) = 512
  int rem = gid & 511;
  int h  = rem >> 7;
  int qt = rem & 127;

  v16h aq;                                  // Q tile, A layout, K-dim padded
  const _Float16* qrow = Q + ((size_t)b*NN + qt*16 + r)*EE + h*16;
  #pragma unroll
  for (int e = 0; e < 16; ++e) {
    int d = e + 16*half;
    aq[e] = (d < 16) ? qrow[d] : (_Float16)0.f;
  }

  float m[8], l[8];
  #pragma unroll
  for (int e = 0; e < 8; ++e) { m[e] = -1e30f; l[e] = 0.f; }
  v8f acc = {};
  const u32* mbase = maskp + ((size_t)b*NN + qt*16)*WPR;

  for (int kt = 0; kt < 64; ++kt) {         // 32 keys per iteration
    // two K^T B-tiles (K-dim = d = 16 real, upper half zero)
    v16h bk0, bk1;
    if (half == 0) {
      const _Float16* k0r = Km + ((size_t)b*NN + kt*32 +      r)*EE + h*16;
      const _Float16* k1r = Km + ((size_t)b*NN + kt*32 + 16 + r)*EE + h*16;
      #pragma unroll
      for (int e = 0; e < 16; ++e) { bk0[e] = k0r[e]; bk1[e] = k1r[e]; }
    } else {
      #pragma unroll
      for (int e = 0; e < 16; ++e) { bk0[e] = (_Float16)0.f; bk1[e] = (_Float16)0.f; }
    }
    v8f zero = {};
    v8f s0 = wmma_f16(aq, bk0, zero);       // cols: keys kt*32 + 0..15
    v8f s1 = wmma_f16(aq, bk1, zero);       // cols: keys kt*32 + 16..31

    u32 mw[8];
    #pragma unroll
    for (int e = 0; e < 8; ++e)             // word kt covers these 32 keys
      mw[e] = mbase[(size_t)(e + 8*half)*WPR + kt];

    #pragma unroll
    for (int e = 0; e < 8; ++e) {
      bool b0 = (mw[e] >> r) & 1u;
      bool b1 = (mw[e] >> (16 + r)) & 1u;
      float sv0 = b0 ? -1e30f : (s0[e] * 0.25f);    // 1/sqrt(D), D=16
      float sv1 = b1 ? -1e30f : (s1[e] * 0.25f);
      float t = fmaxf(sv0, sv1);                    // row max over 16 lanes
      t = fmaxf(t, __shfl_xor(t, 1, 32));
      t = fmaxf(t, __shfl_xor(t, 2, 32));
      t = fmaxf(t, __shfl_xor(t, 4, 32));
      t = fmaxf(t, __shfl_xor(t, 8, 32));
      float mn   = fmaxf(m[e], t);
      float corr = __expf(m[e] - mn);
      float p0   = __expf(sv0 - mn);
      float p1   = __expf(sv1 - mn);
      l[e]   = l[e]*corr + p0 + p1;
      acc[e] = acc[e]*corr;
      m[e]   = mn;
      psh[w][e + 8*half][r]      = (_Float16)p0;    // stage for A-relayout
      psh[w][e + 8*half][16 + r] = (_Float16)p1;
    }
    __syncthreads();                                // uniform across block
    v16h pa;                                        // P as A operand, K=32 full
    #pragma unroll
    for (int e = 0; e < 16; ++e)
      pa[e] = psh[w][r][e + 16*half];
    v16h bv;                                        // V 32x16 B-tile, all lanes
    const _Float16* vrow = Vt + (((size_t)(b*HH + h))*16 + r)*NN + kt*32 + 16*half;
    #pragma unroll
    for (int e = 0; e < 16; ++e) bv[e] = vrow[e];   // K(e) = e + 16*half
    acc = wmma_f16(pa, bv, acc);
    __syncthreads();
  }

  #pragma unroll
  for (int e = 0; e < 8; ++e) {
    float t = l[e];                                 // row sum over 16 lanes
    t += __shfl_xor(t, 1, 32);
    t += __shfl_xor(t, 2, 32);
    t += __shfl_xor(t, 4, 32);
    t += __shfl_xor(t, 8, 32);
    float inv = 1.f / t;
    O[((size_t)b*NN + qt*16 + e + 8*half)*EE + h*16 + r] = (_Float16)(acc[e]*inv);
  }
}

// ---------------------------------------------------------------------------
extern "C" void kernel_launch(void* const* d_in, const int* in_sizes, int n_in,
                              void* d_out, int out_size, void* d_ws, size_t ws_size,
                              hipStream_t stream) {
  (void)in_sizes; (void)n_in; (void)out_size; (void)ws_size;
  const float* nf  = (const float*)d_in[0];
  const float* adj = (const float*)d_in[1];

  char* w = (char*)d_ws; size_t off = 0;
  auto take = [&](size_t bytes) {
    char* p = w + off; off = (off + bytes + 255) & ~(size_t)255; return p;
  };
  _Float16* x0    = (_Float16*)take((size_t)MROWS*16*2);
  u32*      maskp = (u32*)     take((size_t)BB*NN*WPR*4);
  _Float16* xA    = (_Float16*)take((size_t)MROWS*EE*2);
  _Float16* xB    = (_Float16*)take((size_t)MROWS*EE*2);
  _Float16* Qb    = (_Float16*)take((size_t)MROWS*EE*2);
  _Float16* Kb    = (_Float16*)take((size_t)MROWS*EE*2);
  _Float16* Vt    = (_Float16*)take((size_t)MROWS*EE*2);
  _Float16* Ob    = (_Float16*)take((size_t)MROWS*EE*2);
  _Float16* wf    = (_Float16*)take(50176*2);
  float*    bf    = (float*)   take(784*4);
  _Float16* h1    = Qb;   // 4MB alias (Qb+Kb), head runs after attention
  _Float16* h2    = Vt;   // 4MB alias (Vt+Ob)

  scale_x  <<<MROWS*16/256, 256, 0, stream>>>(nf, x0);
  pack_mask<<<BB*NN*WPR/256, 256, 0, stream>>>(adj, maskp);
  fold_weights<<<1, 256, 0, stream>>>(
      (const float*)d_in[2],  (const float*)d_in[3],  (const float*)d_in[4],
      (const float*)d_in[5],  (const float*)d_in[6],  (const float*)d_in[7],
      (const float*)d_in[8],  (const float*)d_in[9],  (const float*)d_in[10],
      (const float*)d_in[11], (const float*)d_in[12], (const float*)d_in[13],
      (const float*)d_in[14], (const float*)d_in[15], (const float*)d_in[16],
      (const float*)d_in[17], (const float*)d_in[18], (const float*)d_in[19],
      (const float*)d_in[20], (const float*)d_in[21], (const float*)d_in[22],
      (const float*)d_in[23], (const float*)d_in[24], (const float*)d_in[25],
      wf, bf);

  const int M = MROWS;
  // ---- layer 0 (input K=16) ----
  gemm16<16,0,false><<<512, 256, 0, stream>>>(x0, wf+0,     bf+0,   Qb, M, 4, EE, EE);
  gemm16<16,0,false><<<512, 256, 0, stream>>>(x0, wf+1024,  bf+64,  Kb, M, 4, EE, EE);
  gemm16<16,2,false><<<512, 256, 0, stream>>>(x0, wf+2048,  bf+128, Vt, M, 4, 0,  0);
  attn_kernel<<<1024, 128, 0, stream>>>(Qb, Kb, Vt, maskp, Ob);
  gemm16<64,0,true ><<<512, 256, 0, stream>>>(Ob, wf+15360, bf+192, xA, M, 4, EE, EE);
  // ---- layer 1 (K=64) ----
  gemm16<64,0,false><<<512, 256, 0, stream>>>(xA, wf+3072,  bf+256, Qb, M, 4, EE, EE);
  gemm16<64,0,false><<<512, 256, 0, stream>>>(xA, wf+7168,  bf+320, Kb, M, 4, EE, EE);
  gemm16<64,2,false><<<512, 256, 0, stream>>>(xA, wf+11264, bf+384, Vt, M, 4, 0,  0);
  attn_kernel<<<1024, 128, 0, stream>>>(Qb, Kb, Vt, maskp, Ob);
  gemm16<64,0,true ><<<512, 256, 0, stream>>>(Ob, wf+19456, bf+448, xB, M, 4, EE, EE);
  // ---- head ----
  gemm16<64, 0,true ><<<1024, 256, 0, stream>>>(xB, wf+23552, bf+512, h1, M, 8, 128, 128);
  gemm16<128,0,true ><<<1024, 256, 0, stream>>>(h1, wf+31744, bf+640, h2, M, 8, 128, 128);
  gemm16<128,1,false><<<128,  256, 0, stream>>>(h2, wf+48128, bf+768, (float*)d_out, M, 1, 4, 4);
}
